// CosinePredictionWihEdge_88622355186219
// MI455X (gfx1250) — matrix operand order repaired and checked by reading, not compiled, not run on hardware
//
#include <hip/hip_runtime.h>

typedef __attribute__((ext_vector_type(16))) _Float16 v16h;
typedef __attribute__((ext_vector_type(8)))  _Float16 v8h;
typedef __attribute__((ext_vector_type(8)))  float    v8f;
typedef __attribute__((ext_vector_type(4)))  float    v4f;

#define D_EMB 128
#define ED    16
#define K1    272      // 2*D + ED
#define K1P   288      // padded to 9*32 for WMMA K-chunks
#define N1    128
#define N2    32
#define WPB   8        // waves per block (256 threads, wave32)
#define X2S   136      // LDS X2 row stride in halves (128 + 8 pad)
#define X3S   36       // LDS X3 row stride in floats (32 + 4 pad)

// ---------------- weight prep: f32 -> f16, transposed, zero padded ----------
__global__ void prep_weights(const float* __restrict__ W1,
                             const float* __restrict__ W2,
                             _Float16* __restrict__ w1t,
                             _Float16* __restrict__ w2t) {
  int i = blockIdx.x * blockDim.x + threadIdx.x;
  const int n1tot = N1 * K1P;   // 36864
  const int n2tot = N2 * D_EMB; // 4096
  if (i < n1tot) {
    int n = i / K1P, k = i % K1P;
    w1t[i] = (k < K1) ? (_Float16)W1[k * N1 + n] : (_Float16)0.0f;
  } else if (i < n1tot + n2tot) {
    int j = i - n1tot;
    int n = j / D_EMB, k = j % D_EMB;
    w2t[j] = (_Float16)W2[k * N2 + n];
  }
}

// 16 raw f32 values that become one 16x32-f16 A fragment slice for this lane
struct RawA { v4f x0, x1, y0, y1; };

__device__ __forceinline__ RawA load_rawA(int c, const float* __restrict__ rs,
                                          const float* __restrict__ rd,
                                          const float* __restrict__ re, int lhi) {
  const int k0 = c * 32 + lhi * 8;
  RawA r;
  if (c < 4) {
    const float* p = rs + k0;
    r.x0 = *(const v4f*)(p);      r.x1 = *(const v4f*)(p + 4);
    r.y0 = *(const v4f*)(p + 16); r.y1 = *(const v4f*)(p + 20);
  } else if (c < 8) {
    const float* p = rd + (k0 - 128);
    r.x0 = *(const v4f*)(p);      r.x1 = *(const v4f*)(p + 4);
    r.y0 = *(const v4f*)(p + 16); r.y1 = *(const v4f*)(p + 20);
  } else {
    const float* p = re + (k0 - 256);           // ef row (16 f32)
    r.x0 = *(const v4f*)(p);      r.x1 = *(const v4f*)(p + 4);
    r.y0 = (v4f){0.f, 0.f, 0.f, 0.f};           // K >= 272 -> zero pad
    r.y1 = (v4f){0.f, 0.f, 0.f, 0.f};
  }
  return r;
}

__device__ __forceinline__ v16h pack_a(const RawA& r) {
  v16h a;
  #pragma unroll
  for (int j = 0; j < 4; ++j) {
    a[j]      = (_Float16)r.x0[j];
    a[4 + j]  = (_Float16)r.x1[j];
    a[8 + j]  = (_Float16)r.y0[j];
    a[12 + j] = (_Float16)r.y1[j];
  }
  return a;
}

// ---------------- fused edge-MLP ----------------
__global__ __launch_bounds__(256) void edge_mlp_kernel(
    const float* __restrict__ h_user, const float* __restrict__ h_item,
    const float* __restrict__ ef_orders, const float* __restrict__ ef_rev,
    const _Float16* __restrict__ w1t, const _Float16* __restrict__ w2t,
    const float* __restrict__ b1, const float* __restrict__ b2,
    const float* __restrict__ W3, const float* __restrict__ b3,
    const int* __restrict__ src_o, const int* __restrict__ dst_o,
    const int* __restrict__ src_r, const int* __restrict__ dst_r,
    float* __restrict__ out, int E)
{
  __shared__ _Float16 ldsX2[WPB][16][X2S];   // 34816 B
  __shared__ float    ldsX3[WPB][16][X3S];   // 18432 B

  const int lane = threadIdx.x & 31;
  const int wid  = threadIdx.x >> 5;
  const int l15  = lane & 15;
  const int lhi  = lane >> 4;              // 0 or 1 (wave half)
  const int tilesPerType = (E + 15) >> 4;
  const int totalTiles = 2 * tilesPerType;
  const int gwave  = blockIdx.x * WPB + wid;
  const int nwaves = gridDim.x * WPB;

  // tile-invariant biases (splat by output column n = l15 + 16*t)
  float bias1[8];
  #pragma unroll
  for (int t = 0; t < 8; ++t) bias1[t] = b1[t * 16 + l15];
  float bias2[2];
  bias2[0] = b2[l15];
  bias2[1] = b2[16 + l15];
  const float b3s = b3[0];

  #pragma clang loop unroll(disable)
  for (int tile = gwave; tile < totalTiles; tile += nwaves) {
    const int etype = (tile >= tilesPerType) ? 1 : 0;
    const int e0 = (tile - etype * tilesPerType) * 16;
    const float* __restrict__ hs   = etype ? h_item : h_user;
    const float* __restrict__ hd   = etype ? h_user : h_item;
    const float* __restrict__ ef   = etype ? ef_rev : ef_orders;
    const int*   __restrict__ srcI = etype ? src_r : src_o;
    const int*   __restrict__ dstI = etype ? dst_r : dst_o;
    float* __restrict__ outp = out + (long)etype * E;

    // lane owns matrix row m = l15 of the 16-edge tile
    const int m = l15;
    int ei = e0 + m; if (ei >= E) ei = E - 1;
    const long srci = srcI[ei];
    const long dsti = dstI[ei];
    const float* rs = hs + srci * (long)D_EMB;
    const float* rd = hd + dsti * (long)D_EMB;
    const float* re = ef + (long)ei * ED;

    // ---- layer 1: [16,288] x [288,128], rolled K-chunk loop, 1 A frag live --
    v8f acc1[8];
    #pragma unroll
    for (int t = 0; t < 8; ++t) {
      #pragma unroll
      for (int j = 0; j < 8; ++j) acc1[t][j] = bias1[t];
    }

    RawA raw = load_rawA(0, rs, rd, re, lhi);
    const _Float16* wbase = w1t + l15 * K1P + lhi * 8;

    #pragma clang loop unroll(disable)
    for (int c = 0; c < 9; ++c) {
      v16h a = pack_a(raw);
      if (c < 8) raw = load_rawA(c + 1, rs, rd, re, lhi);  // prefetch next chunk
      const _Float16* wp = wbase + c * 32;
      #pragma unroll
      for (int t = 0; t < 8; ++t) {
        v8h blo = *(const v8h*)(wp + t * 16 * K1P);
        v8h bhi = *(const v8h*)(wp + t * 16 * K1P + 16);
        v16h bf = __builtin_shufflevector(blo, bhi,
            0, 1, 2, 3, 4, 5, 6, 7, 8, 9, 10, 11, 12, 13, 14, 15);
        acc1[t] = __builtin_amdgcn_wmma_f32_16x16x32_f16(
            false, a, false, bf, (short)0, acc1[t], false, false);
      }
    }

    // ---- relu + transpose D-layout -> row-major f16 via wave-private LDS ----
    _Float16 (*x2)[X2S] = ldsX2[wid];
    #pragma unroll
    for (int t = 0; t < 8; ++t) {
      #pragma unroll
      for (int j = 0; j < 8; ++j) {
        float v = acc1[t][j];
        v = v > 0.f ? v : 0.f;
        x2[lhi * 8 + j][t * 16 + l15] = (_Float16)v;  // row m, col n
      }
    }
    __builtin_amdgcn_fence(__ATOMIC_ACQ_REL, "wavefront"); // DS in-order per wave

    // ---- layer 2: [16,128] x [128,32] via 8 WMMAs ----
    v8f acc2[2];
    #pragma unroll
    for (int t = 0; t < 2; ++t) {
      #pragma unroll
      for (int j = 0; j < 8; ++j) acc2[t][j] = bias2[t];
    }
    #pragma unroll
    for (int c = 0; c < 4; ++c) {
      const int k0 = c * 32 + lhi * 16;
      const _Float16* p = &x2[m][k0];
      v8h lo = *(const v8h*)(p);        // K k0..k0+7
      v8h hi = *(const v8h*)(p + 8);    // K k0+8..k0+15
      v16h a2 = __builtin_shufflevector(lo, hi,
          0, 1, 2, 3, 4, 5, 6, 7, 8, 9, 10, 11, 12, 13, 14, 15);
      const int kb = c * 32 + lhi * 16;
      #pragma unroll
      for (int t = 0; t < 2; ++t) {
        v16h bf = *(const v16h*)(w2t + (t * 16 + l15) * D_EMB + kb);
        acc2[t] = __builtin_amdgcn_wmma_f32_16x16x32_f16(
            false, a2, false, bf, (short)0, acc2[t], false, false);
      }
    }

    // ---- relu -> LDS, then layer 3 (32->1) + sigmoid on lanes 0..15 ----
    float (*x3)[X3S] = ldsX3[wid];
    #pragma unroll
    for (int t = 0; t < 2; ++t) {
      #pragma unroll
      for (int j = 0; j < 8; ++j) {
        float v = acc2[t][j];
        x3[lhi * 8 + j][t * 16 + l15] = v > 0.f ? v : 0.f;
      }
    }
    __builtin_amdgcn_fence(__ATOMIC_ACQ_REL, "wavefront");

    if (lane < 16 && (e0 + lane) < E) {
      float s = b3s;
      #pragma unroll
      for (int q = 0; q < 8; ++q) {
        v4f xv = *(const v4f*)&x3[lane][q * 4];
        v4f wv = *(const v4f*)&W3[q * 4];
        s += xv[0] * wv[0] + xv[1] * wv[1] + xv[2] * wv[2] + xv[3] * wv[3];
      }
      outp[e0 + lane] = 1.0f / (1.0f + __expf(-s));
    }
    __builtin_amdgcn_fence(__ATOMIC_ACQ_REL, "wavefront"); // WAR guard for next tile
  }
}

extern "C" void kernel_launch(void* const* d_in, const int* in_sizes, int n_in,
                              void* d_out, int out_size, void* d_ws, size_t ws_size,
                              hipStream_t stream) {
  const float* h_user    = (const float*)d_in[0];
  const float* h_item    = (const float*)d_in[1];
  const float* ef_orders = (const float*)d_in[2];
  const float* ef_rev    = (const float*)d_in[3];
  const float* W1        = (const float*)d_in[4];
  const float* b1        = (const float*)d_in[5];
  const float* W2        = (const float*)d_in[6];
  const float* b2        = (const float*)d_in[7];
  const float* W3        = (const float*)d_in[8];
  const float* b3        = (const float*)d_in[9];
  const int*   src_o     = (const int*)d_in[10];
  const int*   dst_o     = (const int*)d_in[11];
  const int*   src_r     = (const int*)d_in[12];
  const int*   dst_r     = (const int*)d_in[13];
  float* out = (float*)d_out;
  const int E = in_sizes[10];

  _Float16* w1t = (_Float16*)d_ws;                                  // 128*288 halves
  _Float16* w2t = (_Float16*)((char*)d_ws + (size_t)N1 * K1P * 2);  // 32*128 halves

  const int prepTot = N1 * K1P + N2 * D_EMB;  // 40960
  prep_weights<<<(prepTot + 255) / 256, 256, 0, stream>>>(W1, W2, w1t, w2t);

  const int tiles = 2 * ((E + 15) / 16);
  int blocks = (tiles + WPB - 1) / WPB;
  if (blocks > 1280) blocks = 1280;
  edge_mlp_kernel<<<blocks, 256, 0, stream>>>(
      h_user, h_item, ef_orders, ef_rev, w1t, w2t, b1, b2, W3, b3,
      src_o, dst_o, src_r, dst_r, out, E);
}
